// IntegratorLanguageBlock_75935021793370
// MI455X (gfx1250) — matrix operand order repaired and checked by reading, not compiled, not run on hardware
//
#include <hip/hip_runtime.h>

// ---------------------------------------------------------------------------
// CDNA5 (gfx1250, wave32) implementation of the IntegratorLanguageBlock.
// GEMMs: v_wmma_f32_16x16x32_bf16, 128x256 workgroup tiles double-buffered in
// LDS, filled by the Tensor Data Mover (tensor_load_to_lds / TENSORcnt) when
// available, cooperative vector loads otherwise.
// ---------------------------------------------------------------------------

typedef __bf16 bf16_t;
typedef __attribute__((ext_vector_type(16))) __bf16 v16bf;
typedef __attribute__((ext_vector_type(8)))  __bf16 v8bf;
typedef __attribute__((ext_vector_type(8)))  float  v8f;
typedef __attribute__((ext_vector_type(4)))  unsigned v4u;
typedef __attribute__((ext_vector_type(8)))  int      v8i;
typedef __attribute__((ext_vector_type(4)))  int      v4i;

#if defined(__has_builtin)
#if __has_builtin(__builtin_amdgcn_tensor_load_to_lds) && __has_builtin(__builtin_amdgcn_s_wait_tensorcnt)
#define USE_TDM 1
#endif
#endif
#ifndef USE_TDM
#define USE_TDM 0
#endif

constexpr int Bb   = 4;
constexpr int Ss   = 2048;
constexpr int Dm   = 1024;
constexpr int Hh   = 16;
constexpr int DH   = 64;        // Dm / Hh
constexpr int FFm  = 4096;
constexpr int Mtok = Bb * Ss;   // 8192 tokens
constexpr float DTc = 0.1f;
constexpr int  NITER = 5;

// GEMM tiling
constexpr int BM = 128;         // workgroup tile M
constexpr int BN = 256;         // workgroup tile N
constexpr int BK = 32;          // K slice (one WMMA K)
constexpr int LDK = 40;         // padded LDS row stride in elements (20 dwords)

// ----------------------------- small helpers ------------------------------

__device__ __forceinline__ bf16_t f2bf(float f) {
  unsigned u = __builtin_bit_cast(unsigned, f);
  unsigned r = (u + 0x7FFFu + ((u >> 16) & 1u)) >> 16;   // round-to-nearest-even
  unsigned short s = (unsigned short)r;
  return __builtin_bit_cast(bf16_t, s);
}

__device__ __forceinline__ v8f zero8() {
  v8f z;
#pragma unroll
  for (int i = 0; i < 8; ++i) z[i] = 0.f;
  return z;
}

__device__ __forceinline__ v8f wmma_bf16(v16bf a, v16bf b, v8f c) {
  return __builtin_amdgcn_wmma_f32_16x16x32_bf16(false, a, false, b,
                                                 (short)0, c, false, false);
}

// A-fragment (16x32 MxK) from a row-major [rows, ld] bf16 matrix in GLOBAL.
__device__ __forceinline__ v16bf frag_a(const bf16_t* __restrict__ base,
                                        int row, int ld, int k0, int lane) {
  const int half = lane >> 4;
  const bf16_t* p = base + (size_t)row * ld + k0 + half * 8;
  v8bf lo = *(const v8bf*)p;
  v8bf hi = *(const v8bf*)(p + 16);
  return __builtin_shufflevector(lo, hi, 0,1,2,3,4,5,6,7,8,9,10,11,12,13,14,15);
}

// B-fragment (32x16 KxN) from row-major [N, ld] (NT gemm) in GLOBAL.
__device__ __forceinline__ v16bf frag_b(const bf16_t* __restrict__ base,
                                        int row, int ld, int k0, int lane) {
  const int half = lane >> 4;
  const bf16_t* p = base + (size_t)row * ld + k0 + half * 8;
  v8bf lo = *(const v8bf*)p;
  v8bf hi = *(const v8bf*)(p + 16);
  return __builtin_shufflevector(lo, hi, 0,1,2,3,4,5,6,7,8,9,10,11,12,13,14,15);
}

// Fragment loads from padded LDS tiles (row stride LDK elems = 20 dwords;
// rows 0..15 map to distinct bank groups).
__device__ __forceinline__ v16bf lds_frag_a(const bf16_t* s, int row, int lane) {
  const int half = lane >> 4;
  const bf16_t* p = s + row * LDK + half * 8;
  v8bf lo = *(const v8bf*)p;
  v8bf hi = *(const v8bf*)(p + 16);
  return __builtin_shufflevector(lo, hi, 0,1,2,3,4,5,6,7,8,9,10,11,12,13,14,15);
}
__device__ __forceinline__ v16bf lds_frag_b(const bf16_t* s, int row, int lane) {
  const int half = lane >> 4;
  const bf16_t* p = s + row * LDK + half * 8;
  v8bf lo = *(const v8bf*)p;
  v8bf hi = *(const v8bf*)(p + 16);
  return __builtin_shufflevector(lo, hi, 0,1,2,3,4,5,6,7,8,9,10,11,12,13,14,15);
}

// ----------------------- tile staging (TDM / fallback) --------------------

#if USE_TDM
// Issue one TDM 2D-tile DMA: tile [tileRows x 32] bf16 from row-major
// [totRows, ld] at (row0, k0) into LDS at ldsAddr, writing padded rows
// (16 dwords data + 4 dwords pad = LDK elems). Descriptor per ISA 08 §8.
__device__ __forceinline__ void tdm_tile(const bf16_t* g, int row0, int k0, int ld,
                                         int totRows, int totK,
                                         unsigned ldsAddr, int tileRows) {
  unsigned long long ga = (unsigned long long)(const void*)(g + (size_t)row0 * ld + k0);
  v4u g0;
  g0[0] = 1u;                                            // count=1, user descriptor
  g0[1] = ldsAddr;                                       // lds_addr [63:32]
  g0[2] = (unsigned)(ga & 0xFFFFFFFFu);                  // global_addr low
  g0[3] = (unsigned)((ga >> 32) & 0x01FFFFFFu) | (2u << 30);  // addr hi | type=2
  v8i g1;
  // data_size=1 (2B) | pad_enable | pad_interval=3 (16 dw) | pad_amount=3 (4 dw)
  g1[0] = (int)0x06D10000u;
  g1[1] = (int)(((unsigned)totK & 0xFFFFu) << 16);       // tensor_dim0 lo16
  g1[2] = (int)((((unsigned)totK >> 16) & 0xFFFFu) |
                (((unsigned)totRows & 0xFFFFu) << 16));  // dim0 hi16 | dim1 lo16
  g1[3] = (int)((((unsigned)totRows >> 16) & 0xFFFFu) |
                ((unsigned)BK << 16));                   // dim1 hi16 | tile_dim0=32
  g1[4] = (int)((unsigned)tileRows & 0xFFFFu);           // tile_dim1 | tile_dim2=0
  g1[5] = (int)(unsigned)ld;                             // tensor_dim0_stride lo32
  g1[6] = 0;                                             // stride0 hi16 | stride1 lo16
  g1[7] = 0;
  v4i z4 = {0, 0, 0, 0};
  v8i z8 = {0, 0, 0, 0, 0, 0, 0, 0};
  // 6-arg toolchain form: (g0, g1, g2, g3, extra, cpol)
  __builtin_amdgcn_tensor_load_to_lds(g0, g1, z4, z4, z8, 0);
}
#else
// Cooperative fallback: 256 threads stage the tile with 16B vector copies.
__device__ __forceinline__ void coop_tile(const bf16_t* g, int row0, int k0, int ld,
                                          bf16_t* s, int tileRows, int tid) {
  for (int c = tid; c < tileRows * 4; c += 256) {
    const int r = c >> 2, q = c & 3;
    *(v8bf*)(s + r * LDK + q * 8) =
        *(const v8bf*)(g + (size_t)(row0 + r) * ld + k0 + q * 8);
  }
}
#endif

// ----------------------------- utility kernels ----------------------------

__global__ __launch_bounds__(256) void cvt_bf16(const float* __restrict__ src,
                                                bf16_t* __restrict__ dst, size_t n) {
  size_t i = (size_t)blockIdx.x * 256 + threadIdx.x;
  size_t stride = (size_t)gridDim.x * 256;
  for (; i < n; i += stride) dst[i] = f2bf(src[i]);
}

__global__ __launch_bounds__(256) void ln_rows(const float* __restrict__ x,
                                               const float* __restrict__ w,
                                               const float* __restrict__ b,
                                               bf16_t* __restrict__ ob,
                                               float* __restrict__ of) {
  __shared__ float red[256];
  const int row = blockIdx.x, t = threadIdx.x;
  const float* xr = x + (size_t)row * Dm;
  float s = 0.f;
  for (int c = t; c < Dm; c += 256) s += xr[c];
  red[t] = s; __syncthreads();
  for (int off = 128; off > 0; off >>= 1) { if (t < off) red[t] += red[t + off]; __syncthreads(); }
  const float mu = red[0] * (1.f / Dm);
  __syncthreads();
  float s2 = 0.f;
  for (int c = t; c < Dm; c += 256) { float d = xr[c] - mu; s2 += d * d; }
  red[t] = s2; __syncthreads();
  for (int off = 128; off > 0; off >>= 1) { if (t < off) red[t] += red[t + off]; __syncthreads(); }
  const float rstd = rsqrtf(red[0] * (1.f / Dm) + 1e-5f);
  for (int c = t; c < Dm; c += 256) {
    float y = (xr[c] - mu) * rstd * w[c] + b[c];
    if (ob) ob[(size_t)row * Dm + c] = f2bf(y);
    if (of) of[(size_t)row * Dm + c] = y;
  }
}

__global__ __launch_bounds__(256) void qkv_split(const float* __restrict__ qkv,
                                                 bf16_t* __restrict__ Q,
                                                 bf16_t* __restrict__ K,
                                                 bf16_t* __restrict__ Vt) {
  const int row = blockIdx.x;
  const int b = row / Ss, s = row % Ss;
  const float* src = qkv + (size_t)row * (3 * Dm);
  for (int c = threadIdx.x; c < Dm; c += 256) {
    const int h = c / DH, i = c % DH;
    const size_t bh = (size_t)b * Hh + h;
    Q[(bh * Ss + s) * DH + i]  = f2bf(src[c] * 0.125f);      // 1/sqrt(64) folded
    K[(bh * Ss + s) * DH + i]  = f2bf(src[Dm + c]);
    Vt[(bh * DH + i) * Ss + s] = f2bf(src[2 * Dm + c]);
  }
}

__global__ __launch_bounds__(256) void inl_rows(const float* __restrict__ x1,
                                                const float* __restrict__ w,
                                                const float* __restrict__ b,
                                                const float* __restrict__ u,
                                                const float* __restrict__ al,
                                                const float* __restrict__ be,
                                                const float* __restrict__ g,
                                                float* __restrict__ x2) {
  __shared__ float red[256];
  const int row = blockIdx.x, t = threadIdx.x;
  const float* xr = x1 + (size_t)row * Dm;
  float s = 0.f;
  for (int c = t; c < Dm; c += 256) s += xr[c];
  red[t] = s; __syncthreads();
  for (int off = 128; off > 0; off >>= 1) { if (t < off) red[t] += red[t + off]; __syncthreads(); }
  const float mu = red[0] * (1.f / Dm);
  __syncthreads();
  float s2 = 0.f;
  for (int c = t; c < Dm; c += 256) { float d = xr[c] - mu; s2 += d * d; }
  red[t] = s2; __syncthreads();
  for (int off = 128; off > 0; off >>= 1) { if (t < off) red[t] += red[t + off]; __syncthreads(); }
  const float rstd = rsqrtf(red[0] * (1.f / Dm) + 1e-5f);
  for (int c = t; c < Dm; c += 256) {
    const size_t idx = (size_t)row * Dm + c;
    float xs = (xr[c] - mu) * rstd * w[c] + b[c];
    const float uu = u[idx], av = al[idx], bv = be[idx], gv = g[idx];
    float vs = 0.f;
#pragma unroll
    for (int it = 0; it < NITER; ++it) {
      vs += DTc * (av * (0.f - xs) - bv * vs + uu);   // TARGET == 0
      xs += DTc * gv * vs;
    }
    x2[idx] = xr[c] + xs;
  }
}

// ------------------------------- NT GEMM ----------------------------------
// C[M,N](f32) = A[M,K] * W[N,K]^T (+bias, activation, bf16 out, residual sum).
// 256 threads = 8 waves; workgroup tile BMxBN, wave tile 64x64; K pipelined
// through double-buffered LDS filled by the TDM (wave 0 issues descriptors).
// EPI: 0 none, 1 sigmoid, 2 softplus, 3 exact GELU.

template <int EPI>
__global__ __launch_bounds__(256) void gemm_nt(const bf16_t* __restrict__ A,
                                               const bf16_t* __restrict__ W,
                                               const float* __restrict__ bias,
                                               float* __restrict__ Cf,
                                               bf16_t* __restrict__ Cb,
                                               const float* __restrict__ res,
                                               float* __restrict__ Csum,
                                               int Mdim, int Ndim, int Kdim) {
  __shared__ bf16_t sA[2][BM * LDK];   // 2 * 10 KB
  __shared__ bf16_t sW[2][BN * LDK];   // 2 * 20 KB

  const int tid  = threadIdx.x;
  const int lane = tid & 31;
  const int wid  = tid >> 5;
  const int waveM = (wid >> 2) * 64;   // 0 or 64
  const int waveN = (wid & 3) * 64;    // 0..192
  const int half = lane >> 4, nl = lane & 15;
  const int mBase = blockIdx.y * BM;
  const int nBase = blockIdx.x * BN;
  const int nK = Kdim / BK;

  v8f acc[4][4];
#pragma unroll
  for (int mf = 0; mf < 4; ++mf)
#pragma unroll
    for (int nf = 0; nf < 4; ++nf) acc[mf][nf] = zero8();

  // stage k-slice 0
#if USE_TDM
  if (wid == 0) {
    tdm_tile(A, mBase, 0, Kdim, Mdim, Kdim, (unsigned)(size_t)&sA[0][0], BM);
    tdm_tile(W, nBase, 0, Kdim, Ndim, Kdim, (unsigned)(size_t)&sW[0][0], BN);
    __builtin_amdgcn_s_wait_tensorcnt(0);
  }
#else
  coop_tile(A, mBase, 0, Kdim, &sA[0][0], BM, tid);
  coop_tile(W, nBase, 0, Kdim, &sW[0][0], BN, tid);
#endif
  __syncthreads();

  int cur = 0;
  for (int ks = 0; ks < nK; ++ks) {
    const int nxt = cur ^ 1;
#if USE_TDM
    if (ks + 1 < nK && wid == 0) {     // prefetch next slice via DMA, overlap WMMA
      tdm_tile(A, mBase, (ks + 1) * BK, Kdim, Mdim, Kdim,
               (unsigned)(size_t)&sA[nxt][0], BM);
      tdm_tile(W, nBase, (ks + 1) * BK, Kdim, Ndim, Kdim,
               (unsigned)(size_t)&sW[nxt][0], BN);
    }
#else
    if (ks + 1 < nK) {
      coop_tile(A, mBase, (ks + 1) * BK, Kdim, &sA[nxt][0], BM, tid);
      coop_tile(W, nBase, (ks + 1) * BK, Kdim, &sW[nxt][0], BN, tid);
    }
#endif
    // 16 WMMAs from the current LDS buffers
    v16bf af[4];
#pragma unroll
    for (int mf = 0; mf < 4; ++mf)
      af[mf] = lds_frag_a(&sA[cur][0], waveM + mf * 16 + nl, lane);
#pragma unroll
    for (int nf = 0; nf < 4; ++nf) {
      v16bf bfr = lds_frag_b(&sW[cur][0], waveN + nf * 16 + nl, lane);
#pragma unroll
      for (int mf = 0; mf < 4; ++mf)
        acc[mf][nf] = wmma_bf16(af[mf], bfr, acc[mf][nf]);
    }
#if USE_TDM
    if (ks + 1 < nK && wid == 0) __builtin_amdgcn_s_wait_tensorcnt(0);
#endif
    __syncthreads();
    cur = nxt;
  }

  // epilogue
#pragma unroll
  for (int nf = 0; nf < 4; ++nf) {
    const int col = nBase + waveN + nf * 16 + nl;
    const float bv = bias ? bias[col] : 0.f;
#pragma unroll
    for (int mf = 0; mf < 4; ++mf) {
#pragma unroll
      for (int r = 0; r < 8; ++r) {
        const int rowi = mBase + waveM + mf * 16 + half * 8 + r;
        float v = acc[mf][nf][r] + bv;
        if constexpr (EPI == 1) v = 1.f / (1.f + __expf(-v));
        else if constexpr (EPI == 2) v = (v > 20.f) ? v : log1pf(__expf(v));
        else if constexpr (EPI == 3) v = 0.5f * v * (1.f + erff(v * 0.70710678f));
        const size_t idx = (size_t)rowi * Ndim + col;
        if (Cf) Cf[idx] = v;
        if (Cb) Cb[idx] = f2bf(v);
        if (Csum) Csum[idx] = v + (res ? res[idx] : 0.f);
      }
    }
  }
}

// ---------------------------- Flash attention -----------------------------
// One wave per 16 query rows per (b,h); kv in chunks of 64; online softmax.

__global__ __launch_bounds__(32) void flash_attn(const bf16_t* __restrict__ Q,
                                                 const bf16_t* __restrict__ K,
                                                 const bf16_t* __restrict__ Vt,
                                                 bf16_t* __restrict__ O) {
  __shared__ float  sS[16][68];
  __shared__ bf16_t sP[16][80];
  __shared__ float  rmax[16], rsum[16], rfac[16];

  const int lane = threadIdx.x & 31;
  const int half = lane >> 4, nl = lane & 15;
  const int qb = blockIdx.x * 16;
  const int bh = blockIdx.y;
  const int bidx = bh / Hh, hidx = bh % Hh;

  const bf16_t* Qb_ = Q  + (size_t)bh * Ss * DH;
  const bf16_t* Kb_ = K  + (size_t)bh * Ss * DH;
  const bf16_t* Vb_ = Vt + (size_t)bh * DH * Ss;

  const v16bf qf0 = frag_a(Qb_, qb + nl, DH, 0, lane);
  const v16bf qf1 = frag_a(Qb_, qb + nl, DH, 32, lane);

  v8f acc[4];
#pragma unroll
  for (int n = 0; n < 4; ++n) acc[n] = zero8();
  if (lane < 16) { rmax[lane] = -3.0e38f; rsum[lane] = 0.f; }
  __syncthreads();

  const int nChunks = (qb + 15) / 64 + 1;
  for (int c = 0; c < nChunks; ++c) {
    const int kb = c * 64;
#pragma unroll
    for (int j = 0; j < 4; ++j) {
      v8f s = zero8();
      s = wmma_bf16(qf0, frag_b(Kb_, kb + j * 16 + nl, DH, 0, lane), s);
      s = wmma_bf16(qf1, frag_b(Kb_, kb + j * 16 + nl, DH, 32, lane), s);
      const int kv = kb + j * 16 + nl;
#pragma unroll
      for (int r = 0; r < 8; ++r) {
        const int qrow = qb + half * 8 + r;
        float v = s[r];
        if (kv > qrow) v = -3.0e38f;
        sS[half * 8 + r][j * 16 + nl] = v;
      }
    }
    __syncthreads();
    if (lane < 16) {
      const int r = lane;
      const float mold = rmax[r];
      float cm = mold;
      for (int c2 = 0; c2 < 64; ++c2) cm = fmaxf(cm, sS[r][c2]);
      const float f = __expf(mold - cm);
      float sum = rsum[r] * f;
      for (int c2 = 0; c2 < 64; ++c2) {
        const float pv = __expf(sS[r][c2] - cm);
        sum += pv;
        sP[r][c2] = f2bf(pv);
      }
      rmax[r] = cm; rsum[r] = sum; rfac[r] = f;
    }
    __syncthreads();
#pragma unroll
    for (int n = 0; n < 4; ++n)
#pragma unroll
      for (int r = 0; r < 8; ++r) acc[n][r] *= rfac[half * 8 + r];
#pragma unroll
    for (int ks = 0; ks < 64; ks += 32) {
      v8bf lo = *(const v8bf*)&sP[nl][ks + half * 8];
      v8bf hi = *(const v8bf*)&sP[nl][ks + 16 + half * 8];
      v16bf pa = __builtin_shufflevector(lo, hi, 0,1,2,3,4,5,6,7,8,9,10,11,12,13,14,15);
#pragma unroll
      for (int n = 0; n < 4; ++n) {
        v16bf vb = frag_b(Vb_, n * 16 + nl, Ss, kb + ks, lane);
        acc[n] = wmma_bf16(pa, vb, acc[n]);
      }
    }
    __syncthreads();
  }
#pragma unroll
  for (int n = 0; n < 4; ++n) {
#pragma unroll
    for (int r = 0; r < 8; ++r) {
      const int q = qb + half * 8 + r;
      const float inv = 1.f / rsum[half * 8 + r];
      const size_t row = (size_t)bidx * Ss + q;
      const size_t col = (size_t)hidx * DH + n * 16 + nl;
      O[row * Dm + col] = f2bf(acc[n][r] * inv);
    }
  }
}

// ------------------------------ host driver -------------------------------

extern "C" void kernel_launch(void* const* d_in, const int* in_sizes, int n_in,
                              void* d_out, int out_size, void* d_ws, size_t ws_size,
                              hipStream_t stream) {
  (void)in_sizes; (void)n_in; (void)out_size; (void)ws_size;

  const float* x          = (const float*)d_in[0];
  const float* ln_attn_w  = (const float*)d_in[1];
  const float* ln_attn_b  = (const float*)d_in[2];
  const float* attn_in_w  = (const float*)d_in[3];
  const float* attn_in_b  = (const float*)d_in[4];
  const float* attn_out_w = (const float*)d_in[5];
  const float* attn_out_b = (const float*)d_in[6];
  const float* ln1_w      = (const float*)d_in[7];
  const float* ln1_b      = (const float*)d_in[8];
  const float* ln2_w      = (const float*)d_in[9];
  const float* ln2_b      = (const float*)d_in[10];
  const float* inl_u_w    = (const float*)d_in[11];
  const float* inl_u_b    = (const float*)d_in[12];
  const float* inl_a_w    = (const float*)d_in[13];
  const float* inl_a_b    = (const float*)d_in[14];
  const float* inl_b_w    = (const float*)d_in[15];
  const float* inl_b_b    = (const float*)d_in[16];
  const float* inl_g_w    = (const float*)d_in[17];
  const float* inl_g_b    = (const float*)d_in[18];
  const float* ff1_w      = (const float*)d_in[19];
  const float* ff1_b      = (const float*)d_in[20];
  const float* ff2_w      = (const float*)d_in[21];
  const float* ff2_b      = (const float*)d_in[22];

  char* p = (char*)d_ws;
  auto take = [&](size_t bytes) -> char* {
    char* r = p;
    p += (bytes + 255) & ~(size_t)255;
    return r;
  };

  const size_t MD = (size_t)Mtok * Dm;
  bf16_t* wInbf = (bf16_t*)take((size_t)3 * Dm * Dm * 2);
  bf16_t* wObf  = (bf16_t*)take((size_t)Dm * Dm * 2);
  bf16_t* wUbf  = (bf16_t*)take((size_t)Dm * Dm * 2);
  bf16_t* wAbf  = (bf16_t*)take((size_t)Dm * Dm * 2);
  bf16_t* wBbf  = (bf16_t*)take((size_t)Dm * Dm * 2);
  bf16_t* wGbf  = (bf16_t*)take((size_t)Dm * Dm * 2);
  bf16_t* wF1bf = (bf16_t*)take((size_t)FFm * Dm * 2);
  bf16_t* wF2bf = (bf16_t*)take((size_t)Dm * FFm * 2);
  bf16_t* tokbf = (bf16_t*)take(MD * 2);                 // xn / ctx / h (reuse)
  float*  qkvf  = (float*)take(MD * 3 * 4);              // qkv; later u/alpha/beta
  float*  gbuf  = (float*)take(MD * 4);
  bf16_t* Qb    = (bf16_t*)take(MD * 2);                 // [B*H, S, dh]
  bf16_t* Kb    = (bf16_t*)take(MD * 2);
  bf16_t* Vt    = (bf16_t*)take(MD * 2);                 // [B*H, dh, S]
  bf16_t* attnb = (bf16_t*)take(MD * 2);
  float*  ctx   = (float*)take(MD * 4);
  float*  x1    = (float*)take(MD * 4);
  bf16_t* ffabf = (bf16_t*)take((size_t)Mtok * FFm * 2);
  float* uP = qkvf;
  float* aP = qkvf + MD;
  float* bP = qkvf + 2 * MD;
  float* gP = gbuf;
  float* x2 = (float*)Qb;           // Q/K region dead after flash_attn

  const dim3 cvtG(2048), cvtB(256);
  cvt_bf16<<<cvtG, cvtB, 0, stream>>>(attn_in_w,  wInbf, (size_t)3 * Dm * Dm);
  cvt_bf16<<<cvtG, cvtB, 0, stream>>>(attn_out_w, wObf,  (size_t)Dm * Dm);
  cvt_bf16<<<cvtG, cvtB, 0, stream>>>(inl_u_w,    wUbf,  (size_t)Dm * Dm);
  cvt_bf16<<<cvtG, cvtB, 0, stream>>>(inl_a_w,    wAbf,  (size_t)Dm * Dm);
  cvt_bf16<<<cvtG, cvtB, 0, stream>>>(inl_b_w,    wBbf,  (size_t)Dm * Dm);
  cvt_bf16<<<cvtG, cvtB, 0, stream>>>(inl_g_w,    wGbf,  (size_t)Dm * Dm);
  cvt_bf16<<<cvtG, cvtB, 0, stream>>>(ff1_w,      wF1bf, (size_t)FFm * Dm);
  cvt_bf16<<<cvtG, cvtB, 0, stream>>>(ff2_w,      wF2bf, (size_t)Dm * FFm);

  // --- attention ---
  ln_rows<<<Mtok, 256, 0, stream>>>(x, ln_attn_w, ln_attn_b, tokbf, nullptr);
  gemm_nt<0><<<dim3(3 * Dm / BN, Mtok / BM), 256, 0, stream>>>(
      tokbf, wInbf, attn_in_b, qkvf, nullptr, nullptr, nullptr, Mtok, 3 * Dm, Dm);
  qkv_split<<<Mtok, 256, 0, stream>>>(qkvf, Qb, Kb, Vt);
  flash_attn<<<dim3(Ss / 16, Bb * Hh), 32, 0, stream>>>(Qb, Kb, Vt, attnb);
  gemm_nt<0><<<dim3(Dm / BN, Mtok / BM), 256, 0, stream>>>(
      attnb, wObf, attn_out_b, ctx, nullptr, x, x1, Mtok, Dm, Dm);

  // --- INL controllers ---
  cvt_bf16<<<cvtG, cvtB, 0, stream>>>(ctx, tokbf, MD);
  gemm_nt<0><<<dim3(Dm / BN, Mtok / BM), 256, 0, stream>>>(
      tokbf, wUbf, inl_u_b, uP, nullptr, nullptr, nullptr, Mtok, Dm, Dm);
  gemm_nt<1><<<dim3(Dm / BN, Mtok / BM), 256, 0, stream>>>(
      tokbf, wAbf, inl_a_b, aP, nullptr, nullptr, nullptr, Mtok, Dm, Dm);
  gemm_nt<2><<<dim3(Dm / BN, Mtok / BM), 256, 0, stream>>>(
      tokbf, wBbf, inl_b_b, bP, nullptr, nullptr, nullptr, Mtok, Dm, Dm);
  gemm_nt<1><<<dim3(Dm / BN, Mtok / BM), 256, 0, stream>>>(
      tokbf, wGbf, inl_g_b, gP, nullptr, nullptr, nullptr, Mtok, Dm, Dm);
  inl_rows<<<Mtok, 256, 0, stream>>>(x1, ln1_w, ln1_b, uP, aP, bP, gP, x2);

  // --- feed-forward ---
  ln_rows<<<Mtok, 256, 0, stream>>>(x2, ln2_w, ln2_b, tokbf, nullptr);
  gemm_nt<3><<<dim3(FFm / BN, Mtok / BM), 256, 0, stream>>>(
      tokbf, wF1bf, ff1_b, nullptr, ffabf, nullptr, nullptr, Mtok, FFm, Dm);
  gemm_nt<0><<<dim3(Dm / BN, Mtok / BM), 256, 0, stream>>>(
      ffabf, wF2bf, ff2_b, nullptr, nullptr, x2, (float*)d_out, Mtok, Dm, FFm);
}
